// GCN_2layers_tunning_61357902791184
// MI455X (gfx1250) — matrix operand order repaired and checked by reading, not compiled, not run on hardware
//
#include <hip/hip_runtime.h>

typedef __attribute__((ext_vector_type(16))) __bf16 v16bf;
typedef __attribute__((ext_vector_type(8)))  __bf16 v8bf;
typedef __attribute__((ext_vector_type(8)))  float  v8f;

__device__ __forceinline__ __bf16 f2bf(float f) {
  unsigned u = __float_as_uint(f);
  unsigned r = u + 0x7FFFu + ((u >> 16) & 1u);   // round-to-nearest-even
  unsigned short h = (unsigned short)(r >> 16);
  __bf16 b;
  __builtin_memcpy(&b, &h, sizeof(h));
  return b;
}

// ---------------- elementwise helpers ----------------
__global__ void zero_f(float* p, long long n) {
  long long i = (long long)blockIdx.x * blockDim.x + threadIdx.x;
  if (i < n) p[i] = 0.0f;
}

__global__ void cvt_bf16(const float* __restrict__ s, __bf16* __restrict__ d, long long n) {
  long long i = (long long)blockIdx.x * blockDim.x + threadIdx.x;
  if (i < n) d[i] = f2bf(s[i]);
}

// src[K][Np] fp32 -> dst[Np][K] bf16 (transpose + convert)
__global__ void cvtT_bf16(const float* __restrict__ s, __bf16* __restrict__ d,
                          int K, int Np) {
  long long i = (long long)blockIdx.x * blockDim.x + threadIdx.x;
  if (i >= (long long)K * Np) return;
  int k = (int)(i / Np), n = (int)(i % Np);
  d[(long long)n * K + k] = f2bf(s[i]);
}

// y -= x (fp32), and emit bf16 copy of the result
__global__ void axpy_sub_cvt(float* __restrict__ y, const float* __restrict__ x,
                             __bf16* __restrict__ ybf, long long n) {
  long long i = (long long)blockIdx.x * blockDim.x + threadIdx.x;
  if (i >= n) return;
  float v = y[i] - x[i];
  y[i] = v;
  ybf[i] = f2bf(v);
}

// in-place ReLU + bf16 copy
__global__ void relu_cvt(float* __restrict__ p, __bf16* __restrict__ pbf, long long n) {
  long long i = (long long)blockIdx.x * blockDim.x + threadIdx.x;
  if (i >= n) return;
  float v = fmaxf(p[i], 0.0f);
  p[i] = v;
  pbf[i] = f2bf(v);
}

// ---------------- graph norm ----------------
__global__ void deg_kernel(const long long* src, const float* w, float* deg, int E) {
  int e = blockIdx.x * blockDim.x + threadIdx.x;
  if (e < E) atomicAdd(&deg[(int)src[e]], w[e]);
}

__global__ void norm_kernel(const long long* src, const long long* dst,
                            const float* w, const float* deg, float* nrm, int E) {
  int e = blockIdx.x * blockDim.x + threadIdx.x;
  if (e >= E) return;
  float ds = deg[(int)src[e]], dd = deg[(int)dst[e]];
  float is = ds > 0.0f ? rsqrtf(ds) : 0.0f;
  float id = dd > 0.0f ? rsqrtf(dd) : 0.0f;
  nrm[e] = -is * w[e] * id;
}

// ---------------- SpMM: Y[dst] += coef*nrm[e]*X[src], edge-parallel ----------------
__global__ void spmm_scatter(const long long* src, const long long* dst,
                             const float* nrm, const float* X, float* Y,
                             int E, int F, float coef) {
  long long gid = (long long)blockIdx.x * blockDim.x + threadIdx.x;
  int fchunks = F >> 2;
  long long tot = (long long)E * fchunks;
  if (gid >= tot) return;
  int e = (int)(gid / fchunks);
  int f = (int)(gid % fchunks) << 2;
  float w = coef * nrm[e];
  int s = (int)src[e], d = (int)dst[e];
  const float4 xv = *(const float4*)(X + (long long)s * F + f);
  float* yp = Y + (long long)d * F + f;
  atomicAdd(yp + 0, w * xv.x);
  atomicAdd(yp + 1, w * xv.y);
  atomicAdd(yp + 2, w * xv.z);
  atomicAdd(yp + 3, w * xv.w);
}

// ---------------- WMMA GEMM ----------------
// C[M,Np] (+)= A[M,K] @ Bt[Np,K]^T (+bias).  A, Bt are bf16 in memory; C fp32.
// One wave owns a 64x16 C strip: 4 M-tiles reuse one B fragment -> 4 WMMA / k-step.
__global__ void gemm_wmma_bf16(const __bf16* __restrict__ A, int lda,
                               const __bf16* __restrict__ Bt, int ldbt,
                               float* __restrict__ C, int ldc,
                               int M, int Np, int K,
                               const float* __restrict__ bias, int accum) {
  const int lane = threadIdx.x & 31;
  const int wave = threadIdx.x >> 5;
  const int t64  = blockIdx.x * (blockDim.x >> 5) + wave;
  const int ntn  = Np >> 4;
  const int ntot = (M >> 6) * ntn;
  if (t64 >= ntot) return;                    // wave-uniform guard
  const int m0 = (t64 / ntn) << 6;            // 64-row strip
  const int n0 = (t64 % ntn) << 4;

  // A fragment (16-bit A 16x32 layout): lane -> row, k-offset 0/8
  const int am    = m0 + (lane & 15);
  const int akoff = (lane >> 4) << 3;         // 0 or 8
  // B fragment (16-bit B 32x16 layout): lane -> col, k-offset 0/8
  const int bn    = n0 + (lane & 15);
  const int bkoff = (lane >> 4) << 3;         // 0 or 8
  // C/D layout: VGPR r -> row +r or +8+r, col = n0 + (lane&15)
  const int cmoff = (lane < 16) ? 0 : 8;
  const int cn    = n0 + (lane & 15);

  v8f acc[4];
  if (accum) {
#pragma unroll
    for (int t = 0; t < 4; ++t)
#pragma unroll
      for (int r = 0; r < 8; ++r)
        acc[t][r] = C[(long long)(m0 + (t << 4) + cmoff + r) * ldc + cn];
  } else {
    float bv = bias ? bias[cn] : 0.0f;
#pragma unroll
    for (int t = 0; t < 4; ++t)
#pragma unroll
      for (int r = 0; r < 8; ++r) acc[t][r] = bv;
  }

  const __bf16* a0   = A + (long long)am * lda;
  const __bf16* brow = Bt + (long long)bn * ldbt;

  for (int k0 = 0; k0 < K; k0 += 32) {
    __builtin_prefetch(brow + k0 + 256, 0, 1);           // global_prefetch_b8
    // B: 8 bf16 at k0+bkoff and 8 at k0+16+bkoff -> two 16B loads
    v8bf bl = *(const v8bf*)(brow + k0 + bkoff);
    v8bf bh = *(const v8bf*)(brow + k0 + 16 + bkoff);
    v16bf b = __builtin_shufflevector(bl, bh, 0, 1, 2, 3, 4, 5, 6, 7,
                                      8, 9, 10, 11, 12, 13, 14, 15);
#pragma unroll
    for (int t = 0; t < 4; ++t) {
      const __bf16* arow = a0 + (long long)(t << 4) * lda;
      v8bf al = *(const v8bf*)(arow + k0 + akoff);        // K = k0+akoff..+7
      v8bf ah = *(const v8bf*)(arow + k0 + 16 + akoff);   // K = k0+16+akoff..+7
      v16bf a = __builtin_shufflevector(al, ah, 0, 1, 2, 3, 4, 5, 6, 7,
                                        8, 9, 10, 11, 12, 13, 14, 15);
      acc[t] = __builtin_amdgcn_wmma_f32_16x16x32_bf16(false, a, false, b,
                                                       (short)0, acc[t],
                                                       false, false);
    }
  }

#pragma unroll
  for (int t = 0; t < 4; ++t)
#pragma unroll
    for (int r = 0; r < 8; ++r)
      C[(long long)(m0 + (t << 4) + cmoff + r) * ldc + cn] = acc[t][r];
}

// ---------------- tiny final FC (N=21 not WMMA-tileable) ----------------
__global__ void fc3_kernel(const float* __restrict__ z, const float* __restrict__ w,
                           const float* __restrict__ b, float* __restrict__ out,
                           int M, int K, int Np) {
  int i = blockIdx.x * blockDim.x + threadIdx.x;
  if (i >= M * Np) return;
  int m = i / Np, n = i % Np;
  float s = b[n];
  for (int k = 0; k < K; ++k) s += z[m * K + k] * w[k * Np + n];
  out[i] = s;
}

extern "C" void kernel_launch(void* const* d_in, const int* in_sizes, int n_in,
                              void* d_out, int out_size, void* d_ws, size_t ws_size,
                              hipStream_t stream) {
  const float*     x    = (const float*)d_in[0];
  const long long* ei   = (const long long*)d_in[1];
  const float*     ew   = (const float*)d_in[2];
  const float*     W1   = (const float*)d_in[3];
  const float*     b1   = (const float*)d_in[4];
  const float*     W2   = (const float*)d_in[5];
  const float*     b2   = (const float*)d_in[6];
  const float*     fc1w = (const float*)d_in[7];
  const float*     fc1b = (const float*)d_in[8];
  const float*     fc2w = (const float*)d_in[9];
  const float*     fc2b = (const float*)d_in[10];
  const float*     fc3w = (const float*)d_in[11];
  const float*     fc3b = (const float*)d_in[12];
  float* out = (float*)d_out;

  const int N = 32768, T = 512, C = 64, KCH = 5;
  const int FC1 = 256, FC2 = 128, NCLS = 21, BATCH = 64;
  const int RC = 32768;                       // RES*C, FC1 input width
  const int E = in_sizes[2];
  const long long NT = (long long)N * T;
  const long long NC = (long long)N * C;

  // ---- fp32 scratch ----
  float* ws   = (float*)d_ws;
  float* deg  = ws;                           // N
  float* nrm  = deg + N;                      // E
  float* bufA = nrm + E;                      // N*T
  float* bufB = bufA + NT;                    // N*T
  float* bufC = bufB + NT;                    // N*T
  float* out1 = bufC + NT;                    // N*C
  float* out2 = out1 + NC;                    // N*C
  float* z1   = out2 + NC;                    // BATCH*FC1
  float* z2   = z1 + (long long)BATCH * FC1;  // BATCH*FC2
  // ---- bf16 scratch ----
  __bf16* Abf  = (__bf16*)(z2 + (long long)BATCH * FC2);  // N*T (shared A staging)
  __bf16* W1t  = Abf + NT;                    // KCH * C * T   ([C][T] per k)
  __bf16* W2t  = W1t + (long long)KCH * C * T;// KCH * C * C
  __bf16* fc1t = W2t + (long long)KCH * C * C;// FC1 * RC
  __bf16* fc2t = fc1t + (long long)FC1 * RC;  // FC2 * FC1
  __bf16* z1bf = fc2t + (long long)FC2 * FC1; // BATCH * FC1

  const long long* srcI = ei;
  const long long* dstI = ei + E;

  auto blocks_of = [](long long n) { return dim3((unsigned)((n + 255) / 256)); };

  auto gemm = [&](const __bf16* A, int lda, const __bf16* Bt, int ldbt,
                  float* Cm, int ldc, int M, int Np, int Kd,
                  const float* bias, int accum) {
    int tiles = (M >> 6) * (Np >> 4);         // 64x16 strips
    int nblk = (tiles + 7) / 8;               // 8 waves / 256-thread block
    gemm_wmma_bf16<<<nblk, 256, 0, stream>>>(A, lda, Bt, ldbt, Cm, ldc,
                                             M, Np, Kd, bias, accum);
  };
  auto spmm = [&](const float* X, float* Y, int F, float coef) {
    long long thr = (long long)E * (F >> 2);
    spmm_scatter<<<blocks_of(thr), 256, 0, stream>>>(srcI, dstI, nrm, X, Y, E, F, coef);
  };
  auto cvt = [&](const float* s, __bf16* d, long long n) {
    cvt_bf16<<<blocks_of(n), 256, 0, stream>>>(s, d, n);
  };

  // ---- weight panels: convert + transpose to bf16 (once per call) ----
  for (int k = 0; k < KCH; ++k)
    cvtT_bf16<<<blocks_of((long long)T * C), 256, 0, stream>>>(
        W1 + (long long)k * T * C, W1t + (long long)k * C * T, T, C);
  for (int k = 0; k < KCH; ++k)
    cvtT_bf16<<<blocks_of((long long)C * C), 256, 0, stream>>>(
        W2 + (long long)k * C * C, W2t + (long long)k * C * C, C, C);
  cvtT_bf16<<<blocks_of((long long)RC * FC1), 256, 0, stream>>>(fc1w, fc1t, RC, FC1);
  cvtT_bf16<<<blocks_of((long long)FC1 * FC2), 256, 0, stream>>>(fc2w, fc2t, FC1, FC2);

  // ---- degree + Chebyshev edge norm ----
  zero_f<<<blocks_of(N), 256, 0, stream>>>(deg, N);
  deg_kernel<<<blocks_of(E), 256, 0, stream>>>(srcI, ew, deg, E);
  norm_kernel<<<blocks_of(E), 256, 0, stream>>>(srcI, dstI, ew, deg, nrm, E);

  // ---- ChebConv layer 1 (T=512 -> C=64) ----
  cvt(x, Abf, NT);                                                  // T0 = x
  gemm(Abf, T, W1t + 0LL * C * T, T, out1, C, N, C, T, b1, 0);
  zero_f<<<blocks_of(NT), 256, 0, stream>>>(bufA, NT);
  spmm(x, bufA, T, 1.0f);                                           // T1 = L x
  cvt(bufA, Abf, NT);
  gemm(Abf, T, W1t + 1LL * C * T, T, out1, C, N, C, T, nullptr, 1);
  zero_f<<<blocks_of(NT), 256, 0, stream>>>(bufB, NT);
  spmm(bufA, bufB, T, 2.0f);                                        // T2 = 2L T1 - T0
  axpy_sub_cvt<<<blocks_of(NT), 256, 0, stream>>>(bufB, x, Abf, NT);
  gemm(Abf, T, W1t + 2LL * C * T, T, out1, C, N, C, T, nullptr, 1);
  zero_f<<<blocks_of(NT), 256, 0, stream>>>(bufC, NT);
  spmm(bufB, bufC, T, 2.0f);                                        // T3 = 2L T2 - T1
  axpy_sub_cvt<<<blocks_of(NT), 256, 0, stream>>>(bufC, bufA, Abf, NT);
  gemm(Abf, T, W1t + 3LL * C * T, T, out1, C, N, C, T, nullptr, 1);
  zero_f<<<blocks_of(NT), 256, 0, stream>>>(bufA, NT);
  spmm(bufC, bufA, T, 2.0f);                                        // T4 = 2L T3 - T2
  axpy_sub_cvt<<<blocks_of(NT), 256, 0, stream>>>(bufA, bufB, Abf, NT);
  gemm(Abf, T, W1t + 4LL * C * T, T, out1, C, N, C, T, nullptr, 1);
  relu_cvt<<<blocks_of(NC), 256, 0, stream>>>(out1, Abf, NC);       // h = relu(out1)

  // ---- ChebConv layer 2 (C=64 -> C=64), h = out1 (fp32), Abf = bf16(h) ----
  gemm(Abf, C, W2t + 0LL * C * C, C, out2, C, N, C, C, b2, 0);
  zero_f<<<blocks_of(NC), 256, 0, stream>>>(bufA, NC);
  spmm(out1, bufA, C, 1.0f);
  cvt(bufA, Abf, NC);
  gemm(Abf, C, W2t + 1LL * C * C, C, out2, C, N, C, C, nullptr, 1);
  zero_f<<<blocks_of(NC), 256, 0, stream>>>(bufB, NC);
  spmm(bufA, bufB, C, 2.0f);
  axpy_sub_cvt<<<blocks_of(NC), 256, 0, stream>>>(bufB, out1, Abf, NC);
  gemm(Abf, C, W2t + 2LL * C * C, C, out2, C, N, C, C, nullptr, 1);
  zero_f<<<blocks_of(NC), 256, 0, stream>>>(bufC, NC);
  spmm(bufB, bufC, C, 2.0f);
  axpy_sub_cvt<<<blocks_of(NC), 256, 0, stream>>>(bufC, bufA, Abf, NC);
  gemm(Abf, C, W2t + 3LL * C * C, C, out2, C, N, C, C, nullptr, 1);
  zero_f<<<blocks_of(NC), 256, 0, stream>>>(bufA, NC);
  spmm(bufC, bufA, C, 2.0f);
  axpy_sub_cvt<<<blocks_of(NC), 256, 0, stream>>>(bufA, bufB, Abf, NC);
  gemm(Abf, C, W2t + 4LL * C * C, C, out2, C, N, C, C, nullptr, 1);

  // ---- MLP head: out2 viewed as [BATCH, RC] ----
  cvt(out2, Abf, NC);
  gemm(Abf, RC, fc1t, RC, z1, FC1, BATCH, FC1, RC, fc1b, 0);
  cvt(z1, z1bf, (long long)BATCH * FC1);
  gemm(z1bf, FC1, fc2t, FC1, z2, FC2, BATCH, FC2, FC1, fc2b, 0);
  fc3_kernel<<<blocks_of((long long)BATCH * NCLS), 256, 0, stream>>>(
      z2, fc3w, fc3b, out, BATCH, FC2, NCLS);
}
